// AttentionModel_36112085025311
// MI455X (gfx1250) — matrix operand order repaired and checked by reading, not compiled, not run on hardware
//
#include <hip/hip_runtime.h>
#include <hip/hip_bf16.h>
#include <math.h>

// Problem sizes
#define VV 50001
#define EE 300
#define EP 320      // E padded to multiple of 32 (WMMA K-step)
#define HH 256
#define G4 1024     // 4*H
#define BB 64
#define TT 256
#define H2 512      // 2*H

typedef __attribute__((ext_vector_type(16))) __bf16 v16bf;
typedef __attribute__((ext_vector_type(8)))  float  v8f;

union FragU { v16bf v; unsigned int u[8]; };

#define WMMA_BF16(A_, B_, C_) \
  __builtin_amdgcn_wmma_f32_16x16x32_bf16(false, (A_), false, (B_), (short)0, (C_), false, false)

__device__ __forceinline__ float sigmoidf_(float x) { return 1.0f / (1.0f + __expf(-x)); }

// ---- WMMA fragment loaders (CDNA5 wave32 VGPR layouts per ISA 7.12.2) ----
// A matrix 16x32 bf16, row M = m0 + lane%16; dword v holds K pair:
//   K = k0 + 16*(v>>2) + 8*(lane>>4) + 2*(v&3)   -> two contiguous 16B runs
__device__ __forceinline__ v16bf load_a_bf16(const __bf16* base, int ld, int m0, int k0, int lane) {
  FragU f;
  const __bf16* row = base + (size_t)(m0 + (lane & 15)) * ld;
  int half8 = (lane >> 4) << 3;
#pragma unroll
  for (int v = 0; v < 8; ++v) {
    int k = k0 + ((v >> 2) << 4) + half8 + ((v & 3) << 1);
    f.u[v] = *(const unsigned int*)(row + k);
  }
  return f.v;
}

// B matrix 32x16 bf16 taken from row-major weight W[N][K] (B[k][n] = W[n][k]):
//   N = n0 + lane%16; dword v holds K pair K = k0 + 16*(v>>2) + 8*(lane>>4) + 2*(v&3)
__device__ __forceinline__ v16bf load_b_bf16(const __bf16* W, int ld, int n0, int k0, int lane) {
  FragU f;
  const __bf16* row = W + (size_t)(n0 + (lane & 15)) * ld;
  int half8 = (lane >> 4) << 3;
#pragma unroll
  for (int v = 0; v < 8; ++v) {
    int k = k0 + ((v >> 2) << 4) + half8 + ((v & 3) << 1);
    f.u[v] = *(const unsigned int*)(row + k);
  }
  return f.v;
}

// A fragment built from fp32 source with in-register bf16 conversion
__device__ __forceinline__ v16bf load_a_f32cvt(const float* base, int ld, int m0, int k0, int lane) {
  v16bf r;
  const float* row = base + (size_t)(m0 + (lane & 15)) * ld;
  int half8 = (lane >> 4) << 3;
#pragma unroll
  for (int v = 0; v < 8; ++v) {
    int k = k0 + ((v >> 2) << 4) + half8 + ((v & 3) << 1);
    r[2 * v]     = (__bf16)row[k];
    r[2 * v + 1] = (__bf16)row[k + 1];
  }
  return r;
}

// ---- K0: weight prep (bf16 conversion, padding, bias combine) ----
__global__ void k_prep(const float* wih_f, const float* wih_b,
                       const float* whh_f, const float* whh_b,
                       const float* bih_f, const float* bhh_f,
                       const float* bih_b, const float* bhh_b,
                       const float* qWy, const float* rWy,
                       __bf16* Wihbf, __bf16* Whhbf, __bf16* Wybf, float* biasc) {
  int task = blockIdx.y;
  int i = blockIdx.x * blockDim.x + threadIdx.x;
  if (task <= 1) {                       // Wih -> bf16 [1024][320], K padded with 0
    if (i < G4 * EP) {
      int n = i / EP, k = i - n * EP;
      const float* src = task ? wih_b : wih_f;
      Wihbf[(size_t)task * G4 * EP + i] = (k < EE) ? (__bf16)src[n * EE + k] : (__bf16)0.0f;
    }
  } else if (task <= 3) {                // Whh -> bf16 [1024][256]
    if (i < G4 * HH) {
      const float* src = (task == 3) ? whh_b : whh_f;
      Whhbf[(size_t)(task - 2) * G4 * HH + i] = (__bf16)src[i];
    }
  } else if (task <= 5) {                // Wy -> bf16 [256][512]
    if (i < HH * H2) {
      const float* src = (task == 5) ? rWy : qWy;
      Wybf[(size_t)(task - 4) * HH * H2 + i] = (__bf16)src[i];
    }
  } else {                               // combined bias per direction
    if (i < 2 * G4) {
      int dir = i / G4, n = i - dir * G4;
      biasc[i] = dir ? (bih_b[n] + bhh_b[n]) : (bih_f[n] + bhh_f[n]);
    }
  }
}

// ---- K1: embedding gather -> bf16, padded to EP ----
__global__ void k_embed(const int* Xq, const int* Xr, const float* emb, __bf16* Xbf) {
  int r   = blockIdx.x;            // r = b*T + t
  int seq = blockIdx.y;
  int k   = threadIdx.x;           // 0..319
  const int* X = seq ? Xr : Xq;
  int idx = X[r];
  __bf16 val = (k < EE) ? (__bf16)emb[(size_t)idx * EE + k] : (__bf16)0.0f;
  Xbf[((size_t)seq * BB * TT + r) * EP + k] = val;
}

// ---- K2: time-invariant input GEMM  G[b*T+t][n] = X @ Wih^T + bias ----
// 2 M-tiles per wave: halves redundant B traffic per WMMA.
__global__ __launch_bounds__(256) void k_input_gemm(const __bf16* Xbf, const __bf16* Wihbf,
                                                    const float* biasc, float* Gbuf) {
  int lane = threadIdx.x & 31;
  int wave = threadIdx.x >> 5;
  int combo = blockIdx.z;          // seq*2 + dir
  int seq = combo >> 1, dir = combo & 1;
  const __bf16* X  = Xbf   + (size_t)seq * BB * TT * EP;
  const __bf16* W  = Wihbf + (size_t)dir * G4 * EP;
  const float* bias = biasc + dir * G4;
  float* Gc = Gbuf + (size_t)combo * BB * TT * G4;

  int m0a   = (blockIdx.x * 16 + wave) * 16;      // 64 blocks cover 1024 M-tiles
  int m0b   = m0a + 128;                          // second tile: wave+8
  int nbase = blockIdx.y * 128;                   // 8 N-tiles per wave

  v8f acc[2][8] = {};
#pragma unroll 1
  for (int kf = 0; kf < EP / 32; ++kf) {
    v16bf a0 = load_a_bf16(X, EP, m0a, kf * 32, lane);
    v16bf a1 = load_a_bf16(X, EP, m0b, kf * 32, lane);
#pragma unroll
    for (int nt = 0; nt < 8; ++nt) {
      v16bf b = load_b_bf16(W, EP, nbase + nt * 16, kf * 32, lane);
      acc[0][nt] = WMMA_BF16(a0, b, acc[0][nt]);
      acc[1][nt] = WMMA_BF16(a1, b, acc[1][nt]);
    }
  }
  int half = lane >> 4, nlo = lane & 15;
#pragma unroll
  for (int mi = 0; mi < 2; ++mi) {
    int mbase = mi ? m0b : m0a;
#pragma unroll
    for (int nt = 0; nt < 8; ++nt) {
      int n = nbase + nt * 16 + nlo;
      float bv = bias[n];
#pragma unroll
      for (int e = 0; e < 8; ++e) {
        int r = mbase + e + 8 * half;              // row = b*T + t
        Gc[(size_t)r * G4 + n] = acc[mi][nt][e] + bv;
      }
    }
  }
}

// ---- K3: persistent recurrent LSTM ----
// Batch rows are independent across time: each chain (seq x dir) is split into
// 4 batch-quarters (16 blocks total), zero cross-workgroup communication.
// 512 threads = 16 waves; wave w owns j-cols [16w,16w+16) for ALL 4 gates of
// the block's 16 batch rows -> LSTM cell entirely in registers.
// The wave's Whh fragments (4 gates x 8 k-frags = 256 VGPRs) are REGISTER-
// PERSISTENT across all 256 timesteps: the steady-state inner loop is just
// LDS A-loads + v_wmma, with no global loads on the critical path.
__global__ __launch_bounds__(512) void k_lstm(const float* Gbuf, const __bf16* Whhbf,
                                              const float* h_q, const float* c_q,
                                              const float* h_r, const float* c_r,
                                              float* Ybuf) {
  __shared__ __attribute__((aligned(16))) __bf16 hbuf[16][HH + 8];
  int tid = threadIdx.x;
  int lane = tid & 31, wave = tid >> 5;       // wave = j-tile 0..15
  int blk = blockIdx.x;                       // 0..15
  int chain = blk >> 2, bq = blk & 3;         // chain, batch-quarter
  int seq = chain >> 1, dir = chain & 1;
  const float* Gc = Gbuf + (size_t)chain * BB * TT * G4;
  const __bf16* W = Whhbf + (size_t)dir * G4 * HH;
  const float* h0 = (seq ? h_r : h_q) + (size_t)dir * BB * HH;
  const float* c0 = (seq ? c_r : c_q) + (size_t)dir * BB * HH;
  float* Yc = Ybuf + (size_t)seq * BB * TT * H2;

  int j0 = wave * 16;
  int half = lane >> 4, nlo = lane & 15;
  int j = j0 + nlo;
  int b0 = bq * 16;                           // absolute batch base of block

  // Persistent Whh fragments (one-time L2 read)
  v16bf breg[4][8];
#pragma unroll
  for (int g = 0; g < 4; ++g)
#pragma unroll
    for (int kf = 0; kf < 8; ++kf)
      breg[g][kf] = load_b_bf16(W, HH, g * 256 + j0, kf * 32, lane);

  for (int i = tid; i < 16 * HH; i += 512)
    hbuf[i >> 8][i & 255] = (__bf16)h0[(b0 + (i >> 8)) * HH + (i & 255)];

  float creg[8];
#pragma unroll
  for (int e = 0; e < 8; ++e)
    creg[e] = c0[(b0 + e + 8 * half) * HH + j];
  __syncthreads();

#pragma unroll 1
  for (int s = 0; s < TT; ++s) {
    int t = dir ? (TT - 1 - s) : s;

    v8f acc[4];                               // i/f/g/o patches
#pragma unroll
    for (int g = 0; g < 4; ++g)
#pragma unroll
      for (int e = 0; e < 8; ++e)
        acc[g][e] = Gc[((size_t)(b0 + e + 8 * half) * TT + t) * G4 + g * 256 + j];

    if (s + 1 < TT) {                         // prefetch next 64KB G slab
      int tn = dir ? (TT - 2 - s) : (s + 1);
      int pb = b0 + (tid >> 5);               // 16 rows
      int pn = (tid & 31) * 32;               // 32 lanes * 128B = 4KB/row
      __builtin_prefetch(&Gc[((size_t)pb * TT + tn) * G4 + pn], 0, 1);
    }

    const __bf16* hptr = &hbuf[0][0];
#pragma unroll
    for (int kf = 0; kf < 8; ++kf) {
      v16bf a = load_a_bf16(hptr, HH + 8, 0, kf * 32, lane);
      acc[0] = WMMA_BF16(a, breg[0][kf], acc[0]);
      acc[1] = WMMA_BF16(a, breg[1][kf], acc[1]);
      acc[2] = WMMA_BF16(a, breg[2][kf], acc[2]);
      acc[3] = WMMA_BF16(a, breg[3][kf], acc[3]);
    }
    __syncthreads();                          // all hbuf reads done

#pragma unroll
    for (int e = 0; e < 8; ++e) {
      int bl = e + 8 * half;                  // local row 0..15
      float zi = acc[0][e], zf = acc[1][e], zg = acc[2][e], zo = acc[3][e];
      float cn = sigmoidf_(zf) * creg[e] + sigmoidf_(zi) * tanhf(zg);
      float hn = sigmoidf_(zo) * tanhf(cn);
      creg[e] = cn;
      hbuf[bl][j] = (__bf16)hn;
      Yc[((size_t)(b0 + bl) * TT + t) * H2 + dir * HH + j] = hn;
    }
    __syncthreads();                          // h_t published
  }
}

// ---- K4: mean over T ----
__global__ void k_mean(const float* Ybuf, float* meanb) {
  int seq = blockIdx.y, b = blockIdx.x, d = threadIdx.x;   // 512 threads
  const float* Y = Ybuf + ((size_t)seq * BB + b) * TT * H2;
  float s = 0.0f;
  for (int t = 0; t < TT; ++t) s += Y[(size_t)t * H2 + d];
  meanb[((size_t)seq * BB + b) * H2 + d] = s * (1.0f / TT);
}

// ---- K5: P[b][h] = mean_b @ Wh^T ----
__global__ void k_projP(const float* meanb, const float* qWh, const float* rWh, float* Pbuf) {
  int seq = blockIdx.y, b = blockIdx.x, h = threadIdx.x;   // 256 threads
  const float* Wh = seq ? rWh : qWh;
  const float* mv = meanb + ((size_t)seq * BB + b) * H2;
  float s = 0.0f;
  for (int d = 0; d < H2; ++d) s += mv[d] * Wh[h * H2 + d];
  Pbuf[((size_t)seq * BB + b) * HH + h] = s;
}

// ---- K6: fused WMMA attention scores: s[b,t] = Wa . tanh(Y@Wy^T + P[b]) ----
__global__ __launch_bounds__(256) void k_att_scores(const float* Ybuf, const __bf16* Wybf,
                                                    const float* qWa, const float* rWa,
                                                    const float* Pbuf, float* scores) {
  int lane = threadIdx.x & 31, wave = threadIdx.x >> 5;
  int seq = blockIdx.y;
  const float* Y = Ybuf + (size_t)seq * BB * TT * H2;
  const __bf16* Wy = Wybf + (size_t)seq * HH * H2;
  const float* Wa = seq ? rWa : qWa;
  const float* P  = Pbuf + (size_t)seq * BB * HH;
  float* sc = scores + (size_t)seq * BB * TT;

  int m0 = (blockIdx.x * 8 + wave) * 16;               // 1024 row-tiles total
  int half = lane >> 4, nlo = lane & 15;

  float rowsum[8];
#pragma unroll
  for (int e = 0; e < 8; ++e) rowsum[e] = 0.0f;

#pragma unroll 1
  for (int g2 = 0; g2 < 2; ++g2) {                     // n-tiles in groups of 8
    v8f acc[8];
#pragma unroll
    for (int nt = 0; nt < 8; ++nt) {
      int n = g2 * 128 + nt * 16 + nlo;
#pragma unroll
      for (int e = 0; e < 8; ++e) {
        int r = m0 + e + 8 * half;                     // r = b*T + t
        acc[nt][e] = P[(r >> 8) * HH + n];
      }
    }
#pragma unroll 1
    for (int kf = 0; kf < H2 / 32; ++kf) {
      v16bf a = load_a_f32cvt(Y, H2, m0, kf * 32, lane);
#pragma unroll
      for (int nt = 0; nt < 8; ++nt) {
        v16bf b = load_b_bf16(Wy, H2, g2 * 128 + nt * 16, kf * 32, lane);
        acc[nt] = WMMA_BF16(a, b, acc[nt]);
      }
    }
#pragma unroll
    for (int nt = 0; nt < 8; ++nt) {
      int n = g2 * 128 + nt * 16 + nlo;
      float wa = Wa[n];
#pragma unroll
      for (int e = 0; e < 8; ++e) rowsum[e] += tanhf(acc[nt][e]) * wa;
    }
  }
#pragma unroll
  for (int e = 0; e < 8; ++e) {                        // reduce 16 lanes per half
    float v = rowsum[e];
    v += __shfl_xor(v, 1, 32);
    v += __shfl_xor(v, 2, 32);
    v += __shfl_xor(v, 4, 32);
    v += __shfl_xor(v, 8, 32);
    rowsum[e] = v;
  }
  if (nlo == 0) {
#pragma unroll
    for (int e = 0; e < 8; ++e) sc[m0 + e + 8 * half] = rowsum[e];
  }
}

// ---- K7: softmax over T ----
__global__ void k_softmax(const float* scores, float* alpha) {
  __shared__ float red[TT];
  int seq = blockIdx.y, b = blockIdx.x, t = threadIdx.x;
  const float* sc = scores + ((size_t)seq * BB + b) * TT;
  float v = sc[t];
  red[t] = v; __syncthreads();
  for (int o = TT / 2; o > 0; o >>= 1) { if (t < o) red[t] = fmaxf(red[t], red[t + o]); __syncthreads(); }
  float mx = red[0]; __syncthreads();
  float e = __expf(v - mx);
  red[t] = e; __syncthreads();
  for (int o = TT / 2; o > 0; o >>= 1) { if (t < o) red[t] += red[t + o]; __syncthreads(); }
  alpha[((size_t)seq * BB + b) * TT + t] = e / red[0];
}

// ---- K8: out[b][d] = sum_t alpha * Y ----
__global__ void k_wsum(const float* Ybuf, const float* alpha, float* out) {
  int seq = blockIdx.y, b = blockIdx.x, d = threadIdx.x;   // 512 threads
  const float* Y  = Ybuf  + ((size_t)seq * BB + b) * TT * H2;
  const float* al = alpha + ((size_t)seq * BB + b) * TT;
  float s = 0.0f;
  for (int t = 0; t < TT; ++t) s += al[t] * Y[(size_t)t * H2 + d];
  out[((size_t)seq * BB + b) * H2 + d] = s;               // quote then response
}

extern "C" void kernel_launch(void* const* d_in, const int* in_sizes, int n_in,
                              void* d_out, int out_size, void* d_ws, size_t ws_size,
                              hipStream_t stream) {
  (void)in_sizes; (void)n_in; (void)out_size; (void)ws_size;
  const int*   Xq    = (const int*)d_in[0];
  const int*   Xr    = (const int*)d_in[1];
  const float* h_q   = (const float*)d_in[2];
  const float* c_q   = (const float*)d_in[3];
  const float* h_r   = (const float*)d_in[4];
  const float* c_r   = (const float*)d_in[5];
  const float* emb   = (const float*)d_in[6];
  const float* wih_f = (const float*)d_in[7];
  const float* whh_f = (const float*)d_in[8];
  const float* bih_f = (const float*)d_in[9];
  const float* bhh_f = (const float*)d_in[10];
  const float* wih_b = (const float*)d_in[11];
  const float* whh_b = (const float*)d_in[12];
  const float* bih_b = (const float*)d_in[13];
  const float* bhh_b = (const float*)d_in[14];
  const float* qWy   = (const float*)d_in[15];
  const float* qWh   = (const float*)d_in[16];
  const float* qWa   = (const float*)d_in[17];
  const float* rWy   = (const float*)d_in[18];
  const float* rWh   = (const float*)d_in[19];
  const float* rWa   = (const float*)d_in[20];

  char* ws = (char*)d_ws;
  size_t off = 0;
  __bf16* Wihbf = (__bf16*)(ws + off); off += (size_t)2 * G4 * EP * 2;        // 1.25 MB
  __bf16* Whhbf = (__bf16*)(ws + off); off += (size_t)2 * G4 * HH * 2;        // 1 MB
  __bf16* Wybf  = (__bf16*)(ws + off); off += (size_t)2 * HH * H2 * 2;        // 0.5 MB
  float*  biasc = (float*)(ws + off);  off += (size_t)2 * G4 * 4;             // 8 KB
  __bf16* Xbf   = (__bf16*)(ws + off); off += (size_t)2 * BB * TT * EP * 2;   // 20 MB
  float*  Gbuf  = (float*)(ws + off);  off += (size_t)4 * BB * TT * G4 * 4;   // 256 MB
  float*  Ybuf  = (float*)(ws + off);  off += (size_t)2 * BB * TT * H2 * 4;   // 64 MB
  float*  meanb = (float*)(ws + off);  off += (size_t)2 * BB * H2 * 4;
  float*  Pbuf  = (float*)(ws + off);  off += (size_t)2 * BB * HH * 4;
  float*  scoreb = (float*)(ws + off); off += (size_t)2 * BB * TT * 4;
  float*  alphab = (float*)(ws + off); off += (size_t)2 * BB * TT * 4;

  k_prep<<<dim3(1280, 7, 1), 256, 0, stream>>>(wih_f, wih_b, whh_f, whh_b,
                                               bih_f, bhh_f, bih_b, bhh_b,
                                               qWy, rWy, Wihbf, Whhbf, Wybf, biasc);
  k_embed<<<dim3(BB * TT, 2, 1), EP, 0, stream>>>(Xq, Xr, emb, Xbf);
  k_input_gemm<<<dim3(64, 8, 4), 256, 0, stream>>>(Xbf, Wihbf, biasc, Gbuf);
  k_lstm<<<dim3(16, 1, 1), 512, 0, stream>>>(Gbuf, Whhbf, h_q, c_q, h_r, c_r, Ybuf);
  k_mean<<<dim3(BB, 2, 1), H2, 0, stream>>>(Ybuf, meanb);
  k_projP<<<dim3(BB, 2, 1), HH, 0, stream>>>(meanb, qWh, rWh, Pbuf);
  k_att_scores<<<dim3(128, 2, 1), 256, 0, stream>>>(Ybuf, Wybf, qWa, rWa, Pbuf, scoreb);
  k_softmax<<<dim3(BB, 2, 1), TT, 0, stream>>>(scoreb, alphab);
  k_wsum<<<dim3(BB, 2, 1), H2, 0, stream>>>(Ybuf, alphab, (float*)d_out);
}
